// OrderQKVAttention_18021682774998
// MI455X (gfx1250) — matrix operand order repaired and checked by reading, not compiled, not run on hardware
//
#include <hip/hip_runtime.h>

#define E_DIM 1024
#define S_LEN 2048
#define B_SZ  2
#define H_NUM 16
#define DH    64
#define MTOT  (B_SZ * S_LEN)   // 4096 rows

typedef __attribute__((ext_vector_type(16))) _Float16     v16h;
typedef __attribute__((ext_vector_type(8)))  float        v8f;
typedef __attribute__((ext_vector_type(4)))  unsigned int v4u;
typedef __attribute__((ext_vector_type(4)))  float        v4f;

struct HFrag { v4u lo, hi; };

// A-fragment (16x32 f16): lane half h holds K-runs [8h..8h+7] and [16+8h..23+8h]
__device__ __forceinline__ v16h ldA16(const _Float16* p) {
  HFrag f;
  f.lo = *reinterpret_cast<const v4u*>(p);
  f.hi = *reinterpret_cast<const v4u*>(p + 16);
  return __builtin_bit_cast(v16h, f);
}
// B-fragment (32x16 f16): lane half h holds K-runs [8h..8h+7] and [16+8h..23+8h]
__device__ __forceinline__ v16h ldB16(const _Float16* p) {
  HFrag f;
  f.lo = *reinterpret_cast<const v4u*>(p);
  f.hi = *reinterpret_cast<const v4u*>(p + 16);
  return __builtin_bit_cast(v16h, f);
}
__device__ __forceinline__ v8f wmma16(v16h a, v16h b, v8f c) {
  return __builtin_amdgcn_wmma_f32_16x16x32_f16(false, a, false, b, (short)0, c,
                                                false, false);
}

// ---------------- stage 0: fp32 -> fp16 convert ----------------
__global__ void k_f2h(const float* __restrict__ in, _Float16* __restrict__ out, int n) {
  int i = blockIdx.x * blockDim.x + threadIdx.x;
  if (i < n) out[i] = (_Float16)in[i];
}

// convert + transpose 1024x1024 weight: WT[n][k] = W[k][n]
__global__ void k_wt(const float* __restrict__ W, _Float16* __restrict__ WT) {
  int i = blockIdx.x * blockDim.x + threadIdx.x;   // i = n*1024 + k
  int k = i & (E_DIM - 1);
  int n = i >> 10;
  WT[i] = (_Float16)W[(size_t)k * E_DIM + n];
}

// ---------------- stage 1: fused QKV + H1 GEMM ----------------
// C[4096,1024] = X[4096,1024] @ W + bias (ReLU for z==3). WG = 128x128 tile,
// 4 waves of 64x64; 16 WMMAs per wave per K-step of 32.
__global__ void __launch_bounds__(128) k_gemm(
    const _Float16* __restrict__ X,
    const _Float16* wt0, const _Float16* wt1, const _Float16* wt2, const _Float16* wt3,
    const float* b0, const float* b1, const float* b2, const float* b3,
    _Float16* o0, _Float16* o1, _Float16* o2, _Float16* o3) {
  const int wave = threadIdx.x >> 5;
  const int lane = threadIdx.x & 31;
  const int half = lane >> 4;
  const int ln   = lane & 15;

  const _Float16* WT; const float* bias; _Float16* out; bool relu = false;
  switch (blockIdx.z) {
    case 0:  WT = wt0; bias = b0; out = o0; break;
    case 1:  WT = wt1; bias = b1; out = o1; break;
    case 2:  WT = wt2; bias = b2; out = o2; break;
    default: WT = wt3; bias = b3; out = o3; relu = true; break;
  }
  const int M0 = blockIdx.y * 128 + (wave >> 1) * 64;
  const int N0 = blockIdx.x * 128 + (wave & 1) * 64;

  const v8f zero = {0.f, 0.f, 0.f, 0.f, 0.f, 0.f, 0.f, 0.f};
  v8f acc[4][4];
#pragma unroll
  for (int i = 0; i < 4; ++i)
#pragma unroll
    for (int j = 0; j < 4; ++j) acc[i][j] = zero;

  for (int k0 = 0; k0 < E_DIM; k0 += 32) {
    v16h af[4], bf[4];
#pragma unroll
    for (int mt = 0; mt < 4; ++mt)
      af[mt] = ldA16(&X[(size_t)(M0 + mt * 16 + ln) * E_DIM + k0 + 8 * half]);
#pragma unroll
    for (int nt = 0; nt < 4; ++nt)
      bf[nt] = ldB16(&WT[(size_t)(N0 + nt * 16 + ln) * E_DIM + k0 + 8 * half]);
    if (k0 + 32 < E_DIM) {
      __builtin_prefetch(&X[(size_t)(M0 + ln) * E_DIM + k0 + 32], 0, 3);
      __builtin_prefetch(&WT[(size_t)(N0 + ln) * E_DIM + k0 + 32], 0, 3);
    }
#pragma unroll
    for (int mt = 0; mt < 4; ++mt)
#pragma unroll
      for (int nt = 0; nt < 4; ++nt)
        acc[mt][nt] = wmma16(af[mt], bf[nt], acc[mt][nt]);
  }
  // epilogue: bias (+ReLU), store fp16
#pragma unroll
  for (int mt = 0; mt < 4; ++mt)
#pragma unroll
    for (int nt = 0; nt < 4; ++nt) {
      int col = N0 + nt * 16 + ln;
      float bvv = bias[col];
#pragma unroll
      for (int r = 0; r < 8; ++r) {
        int row = M0 + mt * 16 + r + 8 * half;
        float v = acc[mt][nt][r] + bvv;
        if (relu) v = fmaxf(v, 0.0f);
        out[(size_t)row * E_DIM + col] = (_Float16)v;
      }
    }
}

// ---------------- stage 2: order head o = relu(h1)@Wo2 + bo2 ----------------
__global__ void __launch_bounds__(128) k_ohead(const _Float16* __restrict__ H1,
                                               const float* __restrict__ Wo2,
                                               const float* __restrict__ bo2,
                                               float* __restrict__ o) {
  const int wave = threadIdx.x >> 5, lane = threadIdx.x & 31;
  const int row = blockIdx.x * 4 + wave;
  float a0 = 0.f, a1 = 0.f, a2 = 0.f, a3 = 0.f;
  for (int k = lane; k < E_DIM; k += 32) {
    float hv = (float)H1[(size_t)row * E_DIM + k];
    v4f w = *reinterpret_cast<const v4f*>(&Wo2[k * 4]);
    a0 += hv * w.x; a1 += hv * w.y; a2 += hv * w.z; a3 += hv * w.w;
  }
  for (int m = 16; m >= 1; m >>= 1) {
    a0 += __shfl_xor(a0, m, 32); a1 += __shfl_xor(a1, m, 32);
    a2 += __shfl_xor(a2, m, 32); a3 += __shfl_xor(a3, m, 32);
  }
  if (lane == 0) {
    float* dst = &o[row * 4];
    dst[0] = a0 + bo2[0]; dst[1] = a1 + bo2[1];
    dst[2] = a2 + bo2[2]; dst[3] = a3 + bo2[3];
  }
}

// ---------------- stage 3: per-head V transpose: Vt[bh][d][t] ----------------
__global__ void k_vt(const _Float16* __restrict__ V, _Float16* __restrict__ Vt) {
  int i = blockIdx.x * blockDim.x + threadIdx.x;   // i = (bh*64 + d)*2048 + t
  int t  = i & (S_LEN - 1);
  int d  = (i >> 11) & (DH - 1);
  int bh = i >> 17;
  int b  = bh >> 4, h = bh & 15;
  Vt[i] = V[(size_t)(b * S_LEN + t) * E_DIM + h * DH + d];
}

// ---------------- stage 4: flash attention (no 1/sqrt(Dh) scale) -------------
// wave handles 16 query rows of one (b,h); loop keys in steps of 32.
__global__ void __launch_bounds__(128) k_attn(const _Float16* __restrict__ Q,
                                              const _Float16* __restrict__ K,
                                              const _Float16* __restrict__ Vt,
                                              float* __restrict__ org) {
  __shared__ alignas(16) _Float16 Pbuf[4][16 * 32];
  const int wave = threadIdx.x >> 5, lane = threadIdx.x & 31;
  const int half = lane >> 4, ln = lane & 15;
  const int bh = blockIdx.y, b = bh >> 4, h = bh & 15;
  const int s0 = blockIdx.x * 64 + wave * 16;

  const size_t qbase = (size_t)(b * S_LEN + s0 + ln) * E_DIM + h * DH;
  v16h qf[2];
  qf[0] = ldA16(&Q[qbase + 0  + 8 * half]);
  qf[1] = ldA16(&Q[qbase + 32 + 8 * half]);

  const v8f zero = {0.f, 0.f, 0.f, 0.f, 0.f, 0.f, 0.f, 0.f};
  v8f oacc[4] = {zero, zero, zero, zero};
  float mrow[8], lrow[8];
#pragma unroll
  for (int r = 0; r < 8; ++r) { mrow[r] = -1e30f; lrow[r] = 0.f; }

  _Float16* pb = &Pbuf[wave][0];

  for (int t0 = 0; t0 < S_LEN; t0 += 32) {
    // S = Q(16x64) @ K^T(64x32): 2 t-subtiles x 2 d-steps
    v8f sacc[2] = {zero, zero};
#pragma unroll
    for (int tt = 0; tt < 2; ++tt) {
      size_t kb = (size_t)(b * S_LEN + t0 + tt * 16 + ln) * E_DIM + h * DH;
      v16h bk0 = ldB16(&K[kb + 0  + 8 * half]);
      v16h bk1 = ldB16(&K[kb + 32 + 8 * half]);
      sacc[tt] = wmma16(qf[0], bk0, sacc[tt]);
      sacc[tt] = wmma16(qf[1], bk1, sacc[tt]);
    }
    // online softmax per accumulator row (row = r + 8*half, col = ln)
#pragma unroll
    for (int r = 0; r < 8; ++r) {
      float s0v = sacc[0][r], s1v = sacc[1][r];
      float mx = fmaxf(s0v, s1v);
      for (int m = 1; m < 16; m <<= 1) mx = fmaxf(mx, __shfl_xor(mx, m, 32));
      float mnew  = fmaxf(mrow[r], mx);
      float scale = __expf(mrow[r] - mnew);
      float p0 = __expf(s0v - mnew), p1 = __expf(s1v - mnew);
      float rs = p0 + p1;
      for (int m = 1; m < 16; m <<= 1) rs += __shfl_xor(rs, m, 32);
      lrow[r] = lrow[r] * scale + rs;
      mrow[r] = mnew;
#pragma unroll
      for (int dt = 0; dt < 4; ++dt) oacc[dt][r] = oacc[dt][r] * scale;
      // spill P tile to LDS in row-major (C-layout -> memory)
      pb[(r + 8 * half) * 32 + ln]      = (_Float16)p0;
      pb[(r + 8 * half) * 32 + ln + 16] = (_Float16)p1;
    }
    // DS ops are in-order within a wave; fence for IR ordering too
    asm volatile("s_wait_dscnt 0" ::: "memory");
    // reload P as an A-fragment (16x32)
    v16h pa = ldA16(&pb[ln * 32 + 8 * half]);
    // O += P(16x32) @ V(32x64) over 4 d-subtiles
#pragma unroll
    for (int dt = 0; dt < 4; ++dt) {
      v16h vb = ldB16(&Vt[(size_t)(bh * DH + dt * 16 + ln) * S_LEN + t0 + 8 * half]);
      oacc[dt] = wmma16(pa, vb, oacc[dt]);
    }
  }
  // normalize and write org (fp32)
#pragma unroll
  for (int dt = 0; dt < 4; ++dt)
#pragma unroll
    for (int r = 0; r < 8; ++r) {
      int srow = s0 + r + 8 * half;
      float v = oacc[dt][r] / lrow[r];
      org[(size_t)(b * S_LEN + srow) * E_DIM + h * DH + dt * 16 + ln] = v;
    }
}

// ---------------- stage 5: polynomial order-weighted mixing ----------------
__global__ void k_mix(const float* __restrict__ org, const float* __restrict__ o,
                      float* __restrict__ out, int n) {
  int i = blockIdx.x * blockDim.x + threadIdx.x;
  if (i >= n) return;
  int row = i >> 10;
  float g = org[i];
  const float* ow = &o[row * 4];
  float p = g;
  float acc = p * ow[0];
  p *= g; acc += p * ow[1];
  p *= g; acc += p * ow[2];
  p *= g; acc += p * ow[3];
  out[i] = acc;
}

extern "C" void kernel_launch(void* const* d_in, const int* in_sizes, int n_in,
                              void* d_out, int out_size, void* d_ws, size_t ws_size,
                              hipStream_t stream) {
  (void)in_sizes; (void)n_in; (void)out_size; (void)ws_size;
  const float* x   = (const float*)d_in[0];
  const float* Wq  = (const float*)d_in[1];
  const float* bq  = (const float*)d_in[2];
  const float* Wk  = (const float*)d_in[3];
  const float* bk  = (const float*)d_in[4];
  const float* Wv  = (const float*)d_in[5];
  const float* bv  = (const float*)d_in[6];
  const float* Wo1 = (const float*)d_in[7];
  const float* bo1 = (const float*)d_in[8];
  const float* Wo2 = (const float*)d_in[9];
  const float* bo2 = (const float*)d_in[10];
  float* out = (float*)d_out;

  char* ws = (char*)d_ws;
  size_t off = 0;
  auto alloc = [&](size_t bytes) -> void* {
    void* p = ws + off;
    off += (bytes + 255) & ~(size_t)255;
    return p;
  };
  _Float16* xh   = (_Float16*)alloc((size_t)MTOT * E_DIM * 2);   //  8 MB
  _Float16* wtq  = (_Float16*)alloc((size_t)E_DIM * E_DIM * 2);  //  2 MB
  _Float16* wtk  = (_Float16*)alloc((size_t)E_DIM * E_DIM * 2);
  _Float16* wtv  = (_Float16*)alloc((size_t)E_DIM * E_DIM * 2);
  _Float16* wto  = (_Float16*)alloc((size_t)E_DIM * E_DIM * 2);
  _Float16* qh   = (_Float16*)alloc((size_t)MTOT * E_DIM * 2);   //  8 MB
  _Float16* kh   = (_Float16*)alloc((size_t)MTOT * E_DIM * 2);
  _Float16* vh   = (_Float16*)alloc((size_t)MTOT * E_DIM * 2);
  _Float16* h1h  = (_Float16*)alloc((size_t)MTOT * E_DIM * 2);
  _Float16* vth  = (_Float16*)alloc((size_t)MTOT * E_DIM * 2);
  float*    obuf = (float*)alloc((size_t)MTOT * 4 * 4);          // 64 KB
  float*    orgb = (float*)alloc((size_t)MTOT * E_DIM * 4);      // 16 MB

  const int n1 = MTOT * E_DIM;     // 4,194,304
  const int n2 = E_DIM * E_DIM;    // 1,048,576

  k_f2h<<<n1 / 256, 256, 0, stream>>>(x, xh, n1);
  k_wt<<<n2 / 256, 256, 0, stream>>>(Wq,  wtq);
  k_wt<<<n2 / 256, 256, 0, stream>>>(Wk,  wtk);
  k_wt<<<n2 / 256, 256, 0, stream>>>(Wv,  wtv);
  k_wt<<<n2 / 256, 256, 0, stream>>>(Wo1, wto);

  dim3 gg(E_DIM / 128, MTOT / 128, 4);   // (8, 32, 4)
  k_gemm<<<gg, 128, 0, stream>>>(xh, wtq, wtk, wtv, wto,
                                 bq, bk, bv, bo1, qh, kh, vh, h1h);

  k_ohead<<<MTOT / 4, 128, 0, stream>>>(h1h, Wo2, bo2, obuf);
  k_vt<<<n1 / 256, 256, 0, stream>>>(vh, vth);

  dim3 ga(S_LEN / 64, B_SZ * H_NUM);     // (32, 32)
  k_attn<<<ga, 128, 0, stream>>>(qh, kh, vth, orgb);

  k_mix<<<n1 / 256, 256, 0, stream>>>(orgb, obuf, out, n1);
}
